// GABVNet_55405078119136
// MI455X (gfx1250) — compile-verified, matrix operand order checked by reading, not run on hardware
//
#include <hip/hip_runtime.h>
#include <hip/hip_bf16.h>
#include <math.h>

#define BB  64
#define SEQ 2048
#define HH_ 64
#define NLAYERS 4
#define CG_STEPS 5

typedef _Float16 v16h __attribute__((ext_vector_type(16)));
typedef float    v8f  __attribute__((ext_vector_type(8)));
typedef float    vf4  __attribute__((ext_vector_type(4)));
typedef float    vf2  __attribute__((ext_vector_type(2)));

union H16 { v16h v; vf4 f4[2]; _Float16 h[16]; };

__device__ __forceinline__ float sigmf_(float x)     { return 1.f / (1.f + expf(-x)); }
__device__ __forceinline__ float softplusf_(float x) { return x > 20.f ? x : log1pf(expf(x)); }

__device__ __forceinline__ float blockReduce256(float v, float* red, int tid) {
    red[tid] = v; __syncthreads();
    #pragma unroll
    for (int s = 128; s > 0; s >>= 1) {
        if (tid < s) red[tid] += red[tid + s];
        __syncthreads();
    }
    float r = red[0]; __syncthreads();
    return r;
}

// ---------------------------------------------------------------- init
__global__ void k_init(float* xest, float* accum, _Float16* hw1h, const float* hw1) {
    int i = blockIdx.x * blockDim.x + threadIdx.x;
    for (int idx = i; idx < BB * SEQ * 2; idx += gridDim.x * blockDim.x) xest[idx] = 0.f;
    if (i < 32 * 128) hw1h[i] = (_Float16)hw1[i];
    if (i == 0) accum[0] = 0.f;
}

// ---------------------------------------------------------------- per-batch scalars (reg / gating MLP)
__global__ void k_scalars(const float* gamma, const float* meta,
                          const float* rw1, const float* rb1, const float* rw2, const float* rb2,
                          const float* pw1, const float* pb1, const float* lng, const float* lnb,
                          const float* pw2, const float* pb2, const float* pw3, const float* pb3,
                          float* scal) {
    int b = threadIdx.x;
    if (b >= BB) return;
    float gam = gamma[b];
    float acc = rb2[0];
    for (int j = 0; j < 16; ++j) {
        float hv = fmaxf(gam * rw1[j] + rb1[j], 0.f);
        acc += hv * rw2[j];
    }
    float reg = softplusf_(acc) + 1e-6f;
    float logvol = -logf(reg) * 3.f;

    float h1[64];
    float mu = 0.f;
    for (int o = 0; o < 64; ++o) {
        float s = pb1[o];
        for (int k = 0; k < 6; ++k) s += meta[b * 6 + k] * pw1[o * 6 + k];
        h1[o] = s; mu += s;
    }
    mu *= (1.f / 64.f);
    float var = 0.f;
    for (int o = 0; o < 64; ++o) { float d = h1[o] - mu; var += d * d; }
    var *= (1.f / 64.f);
    float inv = rsqrtf(var + 1e-5f);
    for (int o = 0; o < 64; ++o) h1[o] = (h1[o] - mu) * inv * lng[o] + lnb[o];

    float h2[32];
    for (int o = 0; o < 32; ++o) {
        float s = pb2[o];
        for (int k = 0; k < 64; ++k) s += fmaxf(h1[k], 0.f) * pw2[o * 64 + k];
        h2[o] = fmaxf(s, 0.f);
    }
    float lg[3];
    for (int o = 0; o < 3; ++o) {
        float s = pb3[o];
        for (int k = 0; k < 32; ++k) s += h2[k] * pw3[o * 32 + k];
        lg[o] = s;
    }
    scal[b]        = sigmf_(lg[0]);                   // g_PN
    scal[64 + b]   = softplusf_(lg[1]) + 0.5f;        // g_NL
    float brake    = sigmf_((logvol + 15.f) * 5.f);
    scal[128 + b]  = sigmf_(lg[2]) * brake;           // g_grad
    scal[192 + b]  = reg;                              // lam
}

// ---------------------------------------------------------------- channel diag (fp64 phase)
__global__ void k_chan(const float* theta, const float* bgp, float* heff, float* hh) {
    int i = blockIdx.x * blockDim.x + threadIdx.x;
    if (i >= BB * SEQ) return;
    int b = i / SEQ, n = i % SEQ;
    const double Cinv = 1.0 / 300000000.0;
    double tau = (double)theta[b * 3 + 0] * Cinv;
    double td  = (double)theta[b * 3 + 1] * Cinv;
    double tdd = (double)theta[b * 3 + 2] * Cinv;
    double t   = (double)n / 1.0e10;
    const double TWO_PI = 6.283185307179586476925286766559;
    double phase = TWO_PI * 3.0e11 * (tau + td * t + 0.5 * tdd * t * t);
    float bg = bgp[0];
    float hr = (float)cos(phase) * bg;
    float hi = (float)(-sin(phase)) * bg;
    heff[2 * i] = hr; heff[2 * i + 1] = hi;
    hh[i] = hr * hr + hi * hi;
}

// ---------------------------------------------------------------- feats (resid or raw)
__global__ void k_feats(const float* yr, const float* yi, const float* xest,
                        const float* accum, float* feats) {
    int i = blockIdx.x * blockDim.x + threadIdx.x;
    if (i >= BB * SEQ) return;
    bool rel = (accum[0] * (1.f / (BB * SEQ))) > 0.1f;
    float a = yr[i], c = yi[i];
    float fr = a, fi = c;
    if (rel) {
        float xr = xest[2 * i], xi = xest[2 * i + 1];
        float am = sqrtf(xr * xr + xi * xi) + 1e-6f;
        float nr = xr / am, ni = xi / am;
        fr = a * nr + c * ni;       // (y * conj(x_norm)).re
        fi = c * nr - a * ni;       // (y * conj(x_norm)).im
    }
    feats[2 * i] = fr; feats[2 * i + 1] = fi;
}

// ---------------------------------------------------------------- bidirectional GRU scan (WMMA core)
// block 0 = forward, block 1 = backward. 512 threads = 16 waves.
// gh(64x192) = h(64x64) @ whh^T : 4x12 = 48 WMMA tiles, 3 per wave, K=64 -> 2 chained 16x16x32 f16 WMMAs.
// whh pre-converted to f16 B-operands held in VGPRs; h in LDS (f16); 2 barriers per step.
// feats are software-pipelined one step ahead into registers so the global-load
// latency hides behind the WMMA phase + barrier instead of stalling each step.
__global__ __launch_bounds__(512, 1) void k_gru(
    const float* whh_f, const float* wih_f, const float* bih_f, const float* bhh_f,
    const float* whh_b, const float* wih_b, const float* bih_b, const float* bhh_b,
    const float* feats, _Float16* rnn) {
    extern __shared__ char smem[];
    float*    sgh  = (float*)smem;                        // 64*200 f32 = 51200 B
    _Float16* sh   = (_Float16*)(smem + 64 * 200 * 4);    // 64*72  f16 =  9216 B
    float*    swih = (float*)(smem + 64 * 200 * 4 + 64 * 72 * 2);   // 384
    float*    sbih = swih + 384;                          // 192
    float*    sbhh = sbih + 192;                          // 192   -> total 63488 B

    const int dir = blockIdx.x;
    const float* whh = dir ? whh_b : whh_f;
    const float* wih = dir ? wih_b : wih_f;
    const float* bih = dir ? bih_b : bih_f;
    const float* bhh = dir ? bhh_b : bhh_f;

    const int tid  = threadIdx.x;
    const int wave = tid >> 5, lane = tid & 31;
    const int lh = lane & 15, hi = (lane >> 4) & 1;

    for (int i = tid; i < 384; i += 512) swih[i] = wih[i];
    for (int i = tid; i < 192; i += 512) { sbih[i] = bih[i]; sbhh[i] = bhh[i]; }
    for (int i = tid; i < 64 * 72; i += 512) sh[i] = (_Float16)0.f;

    // preload B operands: Bmat[k][n] = whh[n*64 + k]
    int mt[3], nt[3];
    v16h bop[3][2];
    #pragma unroll
    for (int q = 0; q < 3; ++q) {
        int t = wave + q * 16;
        mt[q] = t / 12; nt[q] = t % 12;
        const float* row = whh + (nt[q] * 16 + lh) * 64;
        #pragma unroll
        for (int ks = 0; ks < 2; ++ks) {
            int k0 = ks * 32 + hi * 8;
            H16 r;
            #pragma unroll
            for (int j = 0; j < 8; ++j) r.h[j]     = (_Float16)row[k0 + j];
            #pragma unroll
            for (int j = 0; j < 8; ++j) r.h[8 + j] = (_Float16)row[k0 + 16 + j];
            bop[q][ks] = r.v;
        }
    }
    __syncthreads();

    const int gb  = tid >> 3;          // batch row for gate phase
    const int gi0 = (tid & 7) * 8;     // hidden base (8 contiguous)
    const float* fbase = feats + ((size_t)gb * SEQ) * 2;

    // prime the feats pipeline for step 0
    vf2 fcur = *(const vf2*)(fbase + (dir ? (SEQ - 1) : 0) * 2);

    for (int s = 0; s < SEQ; ++s) {
        const int tt = dir ? (SEQ - 1 - s) : s;
        // prefetch feats for the next step (register double-buffer)
        const int sn = (s + 1 < SEQ) ? (s + 1) : s;
        const int tn = dir ? (SEQ - 1 - sn) : sn;
        vf2 fnext = *(const vf2*)(fbase + tn * 2);
        // ---- WMMA: gh = h @ whh^T
        #pragma unroll
        for (int q = 0; q < 3; ++q) {
            v8f c = {0.f, 0.f, 0.f, 0.f, 0.f, 0.f, 0.f, 0.f};
            int m = mt[q] * 16 + lh;
            #pragma unroll
            for (int ks = 0; ks < 2; ++ks) {
                int h0 = ks * 32 + hi * 8;
                H16 a;
                a.f4[0] = *(const vf4*)(sh + m * 72 + h0);
                a.f4[1] = *(const vf4*)(sh + m * 72 + h0 + 16);
                c = __builtin_amdgcn_wmma_f32_16x16x32_f16(false, a.v, false, bop[q][ks],
                                                           (short)0, c, false, false);
            }
            int row0 = mt[q] * 16 + hi * 8;
            int col  = nt[q] * 16 + lh;
            #pragma unroll
            for (int r = 0; r < 8; ++r) sgh[(row0 + r) * 200 + col] = c[r];
        }
        __syncthreads();
        // ---- gates: each thread owns (gb, gi0..gi0+7); reads+writes only its own h elems
        {
            float f0 = fcur.x, f1 = fcur.y;
            H16 hv; hv.f4[0] = *(const vf4*)(sh + gb * 72 + gi0);
            H16 st;
            #pragma unroll
            for (int j = 0; j < 8; ++j) {
                int i = gi0 + j;
                float xr_ = f0 * swih[i * 2]           + f1 * swih[i * 2 + 1]           + sbih[i];
                float xz_ = f0 * swih[(64 + i) * 2]    + f1 * swih[(64 + i) * 2 + 1]    + sbih[64 + i];
                float xn_ = f0 * swih[(128 + i) * 2]   + f1 * swih[(128 + i) * 2 + 1]   + sbih[128 + i];
                float ghr = sgh[gb * 200 + i]        + sbhh[i];
                float ghz = sgh[gb * 200 + 64 + i]   + sbhh[64 + i];
                float ghn = sgh[gb * 200 + 128 + i]  + sbhh[128 + i];
                float rg = sigmf_(xr_ + ghr);
                float zg = sigmf_(xz_ + ghz);
                float ng = tanhf(xn_ + rg * ghn);
                float ho = (float)hv.h[j];
                float hn = (1.f - zg) * ng + zg * ho;
                st.h[j] = (_Float16)hn;
            }
            *(vf4*)(sh + gb * 72 + gi0) = st.f4[0];
            *(vf4*)(rnn + (((size_t)gb) * SEQ + tt) * 128 + dir * 64 + gi0) = st.f4[0];
        }
        __syncthreads();
        fcur = fnext;
    }
}

// ---------------------------------------------------------------- head GEMM: relu(rnn @ hw1^T + hb1) (WMMA)
__global__ __launch_bounds__(256, 2) void k_head(const _Float16* rnn, const _Float16* hw1h,
                                                 const float* hb1, _Float16* relu1) {
    const int tid  = threadIdx.x;
    const int wave = tid >> 5, lane = tid & 31;
    const int lh = lane & 15, hi = (lane >> 4) & 1;
    int id = blockIdx.x * 8 + wave;
    int mtile = id >> 1, ntile = id & 1;
    const _Float16* arow = rnn  + ((size_t)(mtile * 16 + lh)) * 128;
    const _Float16* brow = hw1h + (ntile * 16 + lh) * 128;
    v8f c = {0.f, 0.f, 0.f, 0.f, 0.f, 0.f, 0.f, 0.f};
    #pragma unroll
    for (int ks = 0; ks < 4; ++ks) {
        int h0 = ks * 32 + hi * 8;
        H16 a, bm;
        a.f4[0]  = *(const vf4*)(arow + h0);
        a.f4[1]  = *(const vf4*)(arow + h0 + 16);
        bm.f4[0] = *(const vf4*)(brow + h0);
        bm.f4[1] = *(const vf4*)(brow + h0 + 16);
        c = __builtin_amdgcn_wmma_f32_16x16x32_f16(false, a.v, false, bm.v,
                                                   (short)0, c, false, false);
    }
    int col = ntile * 16 + lh;
    float bb = hb1[col];
    #pragma unroll
    for (int r = 0; r < 8; ++r) {
        int row = mtile * 16 + hi * 8 + r;
        relu1[(size_t)row * 32 + col] = (_Float16)fmaxf(c[r] + bb, 0.f);
    }
}

// ---------------------------------------------------------------- derot / tanh / Onsager / b = conj(h)*z_corr
__global__ __launch_bounds__(256, 2) void k_apply(
    const _Float16* relu1, const float* hw2, const float* hb2, const float* pbias,
    const float* yr, const float* yi, const float* heff, const float* xest,
    const float* scal, const float* ss, float* bvec) {
    __shared__ float shw2[64];
    __shared__ float red[256];
    const int b = blockIdx.x, tid = threadIdx.x;
    if (tid < 64) shw2[tid] = hw2[tid];
    __syncthreads();
    float gpn  = scal[b];
    float beta = ss[0] * scal[64 + b];
    float pb   = pbias[0];
    float br_ = cosf(pb), bi_ = -sinf(pb);
    float zr[8], zi[8], xr[8], xi[8];
    float osum = 0.f;
    #pragma unroll
    for (int j = 0; j < 8; ++j) {
        int n = tid + j * 256;
        size_t i = (size_t)b * SEQ + n;
        const _Float16* rp = relu1 + i * 32;
        float cp = hb2[0], sp = hb2[1];
        #pragma unroll
        for (int k = 0; k < 32; ++k) {
            float rv = (float)rp[k];
            cp += rv * shw2[k];
            sp += rv * shw2[32 + k];
        }
        float mag = sqrtf(cp * cp + sp * sp + 1e-8f);
        float dr = cp / mag, di = -sp / mag;
        float tr = dr * br_ - di * bi_;
        float ti = dr * bi_ + di * br_;
        float er = 1.f - gpn + gpn * tr;
        float ei = gpn * ti;
        float en = sqrtf(er * er + ei * ei) + 1e-8f;
        er /= en; ei /= en;
        float a = yr[i], c = yi[i];
        float zir = a * er - c * ei;
        float zii = a * ei + c * er;
        float tzr = tanhf(beta * zir), tzi = tanhf(beta * zii);
        zr[j] = tzr; zi[j] = tzi;
        xr[j] = xest[2 * i]; xi[j] = xest[2 * i + 1];
        osum += beta * (1.f - tzr * tzr) + beta * (1.f - tzi * tzi);
    }
    float ons = blockReduce256(osum, red, tid) * (0.5f / SEQ);
    #pragma unroll
    for (int j = 0; j < 8; ++j) {
        int n = tid + j * 256;
        size_t i = (size_t)b * SEQ + n;
        float zcr = zr[j] - ons * xr[j];
        float zci = zi[j] - ons * xi[j];
        float hr = heff[2 * i], hIm = heff[2 * i + 1];
        bvec[2 * i]     = hr * zcr + hIm * zci;     // conj(h) * z_corr
        bvec[2 * i + 1] = hr * zci - hIm * zcr;
    }
}

__global__ void k_zero(float* accum) { if (threadIdx.x == 0 && blockIdx.x == 0) accum[0] = 0.f; }

// ---------------------------------------------------------------- diagonal CG + gated update + |x| sum
__global__ __launch_bounds__(256, 2) void k_cg(const float* hh, const float* bvec,
                                               const float* scal, float* xest, float* accum) {
    __shared__ float red[256];
    const int b = blockIdx.x, tid = threadIdx.x;
    float lam = scal[192 + b];
    float gg  = scal[128 + b];
    float da[8], xr[8], xi[8], rr[8], ri[8], pr[8], pi[8], xor_[8], xoi_[8];
    float rs = 0.f;
    #pragma unroll
    for (int j = 0; j < 8; ++j) {
        int n = tid + j * 256;
        size_t i = (size_t)b * SEQ + n;
        da[j] = hh[i] + lam;
        float bvr = bvec[2 * i], bvi = bvec[2 * i + 1];
        xor_[j] = xest[2 * i]; xoi_[j] = xest[2 * i + 1];
        xr[j] = xor_[j]; xi[j] = xoi_[j];
        rr[j] = bvr - da[j] * xr[j];
        ri[j] = bvi - da[j] * xi[j];
        pr[j] = rr[j]; pi[j] = ri[j];
        rs += rr[j] * rr[j] + ri[j] * ri[j];
    }
    rs = blockReduce256(rs, red, tid);
    for (int it = 0; it < CG_STEPS; ++it) {
        float pap = 0.f;
        #pragma unroll
        for (int j = 0; j < 8; ++j) pap += da[j] * (pr[j] * pr[j] + pi[j] * pi[j]);
        pap = blockReduce256(pap, red, tid);
        float alpha = rs / (pap + 1e-20f);
        float rsn = 0.f;
        #pragma unroll
        for (int j = 0; j < 8; ++j) {
            xr[j] += alpha * pr[j]; xi[j] += alpha * pi[j];
            rr[j] -= alpha * da[j] * pr[j]; ri[j] -= alpha * da[j] * pi[j];
            rsn += rr[j] * rr[j] + ri[j] * ri[j];
        }
        rsn = blockReduce256(rsn, red, tid);
        float bta = rsn / (rs + 1e-20f);
        #pragma unroll
        for (int j = 0; j < 8; ++j) { pr[j] = rr[j] + bta * pr[j]; pi[j] = ri[j] + bta * pi[j]; }
        rs = rsn;
    }
    float asum = 0.f;
    #pragma unroll
    for (int j = 0; j < 8; ++j) {
        int n = tid + j * 256;
        size_t i = (size_t)b * SEQ + n;
        float nx = (1.f - gg) * xor_[j] + gg * xr[j];
        float ny = (1.f - gg) * xoi_[j] + gg * xi[j];
        xest[2 * i] = nx; xest[2 * i + 1] = ny;
        asum += sqrtf(nx * nx + ny * ny);
    }
    asum = blockReduce256(asum, red, tid);
    if (tid == 0) atomicAdd(accum, asum);
}

// ---------------------------------------------------------------- output copy
__global__ void k_out(const float* xest, float* out) {
    int i = blockIdx.x * blockDim.x + threadIdx.x;
    if (i < BB * SEQ * 2) out[i] = xest[i];
}

// ---------------------------------------------------------------- launch
extern "C" void kernel_launch(void* const* d_in, const int* in_sizes, int n_in,
                              void* d_out, int out_size, void* d_ws, size_t ws_size,
                              hipStream_t stream) {
    (void)in_sizes; (void)n_in; (void)out_size; (void)ws_size;
    const float* y_real     = (const float*)d_in[0];
    const float* y_imag     = (const float*)d_in[1];
    const float* theta      = (const float*)d_in[2];
    const float* meta       = (const float*)d_in[3];
    const float* gamma_eff  = (const float*)d_in[4];
    const float* scale_slope= (const float*)d_in[5];
    const float* bussgang   = (const float*)d_in[6];
    const float* reg_w1     = (const float*)d_in[7];
    const float* reg_b1     = (const float*)d_in[8];
    const float* reg_w2     = (const float*)d_in[9];
    const float* reg_b2     = (const float*)d_in[10];
    const float* pw1        = (const float*)d_in[11];
    const float* pb1        = (const float*)d_in[12];
    const float* ln_g       = (const float*)d_in[13];
    const float* ln_b       = (const float*)d_in[14];
    const float* pw2        = (const float*)d_in[15];
    const float* pb2        = (const float*)d_in[16];
    const float* pw3        = (const float*)d_in[17];
    const float* pb3        = (const float*)d_in[18];
    const float* gru_wih_f  = (const float*)d_in[19];
    const float* gru_whh_f  = (const float*)d_in[20];
    const float* gru_bih_f  = (const float*)d_in[21];
    const float* gru_bhh_f  = (const float*)d_in[22];
    const float* gru_wih_b  = (const float*)d_in[23];
    const float* gru_whh_b  = (const float*)d_in[24];
    const float* gru_bih_b  = (const float*)d_in[25];
    const float* gru_bhh_b  = (const float*)d_in[26];
    const float* hw1        = (const float*)d_in[27];
    const float* hb1        = (const float*)d_in[28];
    const float* hw2        = (const float*)d_in[29];
    const float* hb2        = (const float*)d_in[30];
    const float* phase_bias = (const float*)d_in[31];

    char* ws = (char*)d_ws;
    size_t off = 0;
    auto alloc = [&](size_t bytes) -> char* {
        char* p = ws + off;
        off = (off + bytes + 255) & ~(size_t)255;
        return p;
    };
    float*    scal  = (float*)alloc(256 * sizeof(float));
    float*    accum = (float*)alloc(sizeof(float));
    _Float16* hw1h  = (_Float16*)alloc(32 * 128 * sizeof(_Float16));
    float*    heff  = (float*)alloc((size_t)BB * SEQ * 2 * sizeof(float));
    float*    hh    = (float*)alloc((size_t)BB * SEQ * sizeof(float));
    float*    xest  = (float*)alloc((size_t)BB * SEQ * 2 * sizeof(float));
    float*    feats = (float*)alloc((size_t)BB * SEQ * 2 * sizeof(float));
    float*    bvec  = (float*)alloc((size_t)BB * SEQ * 2 * sizeof(float));
    _Float16* rnn   = (_Float16*)alloc((size_t)BB * SEQ * 128 * sizeof(_Float16));
    _Float16* relu1 = (_Float16*)alloc((size_t)BB * SEQ * 32 * sizeof(_Float16));

    const int EW_GRID = (BB * SEQ + 255) / 256;
    const size_t gruSmem = 64 * 200 * 4 + 64 * 72 * 2 + (384 + 192 + 192) * 4; // 63488 B

    k_init<<<512, 256, 0, stream>>>(xest, accum, hw1h, hw1);
    k_scalars<<<1, 64, 0, stream>>>(gamma_eff, meta, reg_w1, reg_b1, reg_w2, reg_b2,
                                    pw1, pb1, ln_g, ln_b, pw2, pb2, pw3, pb3, scal);
    k_chan<<<EW_GRID, 256, 0, stream>>>(theta, bussgang, heff, hh);

    for (int layer = 0; layer < NLAYERS; ++layer) {
        k_feats<<<EW_GRID, 256, 0, stream>>>(y_real, y_imag, xest, accum, feats);
        k_gru<<<2, 512, gruSmem, stream>>>(gru_whh_f, gru_wih_f, gru_bih_f, gru_bhh_f,
                                           gru_whh_b, gru_wih_b, gru_bih_b, gru_bhh_b,
                                           feats, rnn);
        k_head<<<2048, 256, 0, stream>>>(rnn, hw1h, hb1, relu1);
        k_apply<<<64, 256, 0, stream>>>(relu1, hw2, hb2, phase_bias,
                                        y_real, y_imag, heff, xest, scal, scale_slope, bvec);
        k_zero<<<1, 1, 0, stream>>>(accum);
        k_cg<<<64, 256, 0, stream>>>(hh, bvec, scal, xest, accum);
    }
    k_out<<<(BB * SEQ * 2 + 255) / 256, 256, 0, stream>>>(xest, (float*)d_out);
}
